// PointNet2PartSeg_64063732187330
// MI455X (gfx1250) — matrix operand order repaired and checked by reading, not compiled, not run on hardware
//
#include <hip/hip_runtime.h>
#include <math.h>

// ---------------------------------------------------------------------------
// PointNet++ part-seg forward for MI455X (gfx1250, wave32, WMMA).
// Dense MLP math -> v_wmma_f32_16x16x32_f16 (f32 accumulate).
// Weights are pre-swizzled into A-fragment order so each lane's fragment is
// one contiguous 32B load; LDS activations are column-major so each lane's
// B fragment is two ds_load_b128.
// ---------------------------------------------------------------------------

typedef _Float16 v16h __attribute__((ext_vector_type(16)));
typedef _Float16 v8h  __attribute__((ext_vector_type(8)));
typedef float    v8f  __attribute__((ext_vector_type(8)));

#define B_  16
#define N_  2048
#define S1_ 512
#define S2_ 128
#define NS_ 64      // nsample (ball query group size)

// ---------------------------------------------------------------------------
// Weight repack: f32 (O x C) row-major -> f16 fragment-order, zero padded.
// Fragment layout (contiguous k order used here, 16-bit A 16x32):
//   lane<16: row=lane,     elem i -> K=i
//   lane>=16: row=lane-16, K += 16
// dst index = ((to*nkb + kblk)*32 + lane)*16 + i
// ---------------------------------------------------------------------------
__global__ void k_repack_w(const float* __restrict__ src, _Float16* __restrict__ dst,
                           int O, int C, int nkb, int total)
{
    int t = blockIdx.x * blockDim.x + threadIdx.x;
    if (t >= total) return;
    int i    = t & 15;
    int lane = (t >> 4) & 31;
    int blk  = t >> 9;
    int kblk = blk % nkb, to = blk / nkb;
    int row  = to * 16 + (lane & 15);
    int ka   = (lane >= 16) ? 16 : 0;
    int kk   = kblk * 32 + ka + i;
    float v  = (row < O && kk < C) ? src[(size_t)row * C + kk] : 0.0f;
    dst[t] = (_Float16)v;
}

// One 16x16x32 WMMA step: A fragment = contiguous 32B from swizzled weights,
// B fragment = contiguous 32B from a column-major LDS activation column.
__device__ __forceinline__ v8f wmma_step(const v16h* __restrict__ Wv, int fragIdx,
                                         const _Float16* Xcol, int lane, v8f acc)
{
    v16h a  = Wv[fragIdx * 32 + lane];
    v16h bf = *(const v16h*)(Xcol + ((lane >= 16) ? 16 : 0));
    return __builtin_amdgcn_wmma_f32_16x16x32_f16(false, a, false, bf,
                                                  (short)0, acc, false, false);
}

// One MLP layer fully in LDS (column-major, strideIn/strideOut halves per col):
// Xout(O x 64) = relu((W*Xin + b)*s + t).  8 waves cover all 16x16 tiles.
__device__ void mlp_layer_lds(const _Float16* Xin, int strideIn,
                              _Float16* Xout, int strideOut,
                              const _Float16* __restrict__ Wswz,
                              const float* __restrict__ bias,
                              const float* __restrict__ sc,
                              const float* __restrict__ tt,
                              int Cpad, int O, int tid)
{
    const int lane = tid & 31, wave = tid >> 5;
    const int nkb = Cpad >> 5;
    const int totalTiles = (O >> 4) * 4;          // M = 64 -> 4 col tiles
    const v16h* Wv = (const v16h*)Wswz;
    for (int tile = wave; tile < totalTiles; tile += 8) {
        const int to = tile >> 2, tm = tile & 3;
        const int col = tm * 16 + (lane & 15);
        const _Float16* Xc = Xin + (size_t)col * strideIn;
        v8f acc = {};
        for (int kb = 0; kb < nkb; ++kb)
            acc = wmma_step(Wv, to * nkb + kb, Xc + kb * 32, lane, acc);
        const int rb = to * 16 + ((lane >= 16) ? 8 : 0);
        v8h o8;
#pragma unroll
        for (int r = 0; r < 8; ++r) {
            int o = rb + r;
            float v = (acc[r] + bias[o]) * sc[o] + tt[o];
            o8[r] = (_Float16)fmaxf(v, 0.0f);
        }
        *(v8h*)(Xout + (size_t)col * strideOut + rb) = o8;
    }
}

// ---------------------------------------------------------------------------
// (B,3,N) -> (B,N,3) packed points
// ---------------------------------------------------------------------------
__global__ void k_pack_xyz(const float* __restrict__ xyz, float* __restrict__ pts)
{
    int t = blockIdx.x * blockDim.x + threadIdx.x;
    if (t >= B_ * N_) return;
    int b = t / N_, n = t % N_;
    pts[(size_t)t * 3 + 0] = xyz[((size_t)b * 3 + 0) * N_ + n];
    pts[(size_t)t * 3 + 1] = xyz[((size_t)b * 3 + 1) * N_ + n];
    pts[(size_t)t * 3 + 2] = xyz[((size_t)b * 3 + 2) * N_ + n];
}

// ---------------------------------------------------------------------------
// Farthest point sampling: one block per batch, dist array in LDS.
// Matches jnp.argmax first-max tie-break (lowest index wins).
// ---------------------------------------------------------------------------
__global__ void __launch_bounds__(256) k_fps(const float* __restrict__ pts,
                                             float* __restrict__ ctr,
                                             int Np, int S)
{
    const int b = blockIdx.x, tid = threadIdx.x;
    __shared__ float dist[2048];
    __shared__ float rM[256];
    __shared__ int   rI[256];
    __shared__ float cc[3];
    for (int i = tid; i < Np; i += 256) dist[i] = 1e10f;
    int far = 0;
    for (int it = 0; it < S; ++it) {
        if (tid == 0) {
            const float* p = pts + ((size_t)b * Np + far) * 3;
            cc[0] = p[0]; cc[1] = p[1]; cc[2] = p[2];
            float* o = ctr + ((size_t)b * S + it) * 3;
            o[0] = p[0]; o[1] = p[1]; o[2] = p[2];
        }
        __syncthreads();
        const float cx = cc[0], cy = cc[1], cz = cc[2];
        float bestd = -1.0f; int besti = 0;
        for (int i = tid; i < Np; i += 256) {
            const float* p = pts + ((size_t)b * Np + i) * 3;
            float dx = p[0] - cx, dy = p[1] - cy, dz = p[2] - cz;
            float nd = fminf(dist[i], dx * dx + dy * dy + dz * dz);
            dist[i] = nd;
            if (nd > bestd) { bestd = nd; besti = i; }
        }
        rM[tid] = bestd; rI[tid] = besti;
        __syncthreads();
        for (int off = 128; off > 0; off >>= 1) {
            if (tid < off) {
                float om = rM[tid + off]; int oi = rI[tid + off];
                if (om > rM[tid] || (om == rM[tid] && oi < rI[tid])) {
                    rM[tid] = om; rI[tid] = oi;
                }
            }
            __syncthreads();
        }
        far = rI[0];
        __syncthreads();
    }
}

// ---------------------------------------------------------------------------
// Fused set-abstraction: ball query (first NS_ hits in index order, padded
// with first hit), relative-coord grouping + feature gather, 3 WMMA MLP
// layers in LDS (column-major), max-pool. One workgroup per (s,b).
// ---------------------------------------------------------------------------
__global__ void __launch_bounds__(256) k_sa_fused(
    const float* __restrict__ pts,    // (B,Np,3)
    const float* __restrict__ feats,  // (B,Cf,Np) or nullptr
    const float* __restrict__ ctr,    // (B,S,3)
    int Np, int S, int Cf, float r2,
    const _Float16* __restrict__ W1, const float* __restrict__ b1,
    const float* __restrict__ s1, const float* __restrict__ t1, int C1,
    const _Float16* __restrict__ W2, const float* __restrict__ b2,
    const float* __restrict__ s2, const float* __restrict__ t2, int C2,
    const _Float16* __restrict__ W3, const float* __restrict__ b3,
    const float* __restrict__ s3, const float* __restrict__ t3, int C3,
    int C0, int C0pad,
    float* __restrict__ out)          // (B,C3,S)
{
    const int s = blockIdx.x, b = blockIdx.y, tid = threadIdx.x;

    // column-major activations: shA = 64 cols x 160 rows, shB = 64 x 256
    __shared__ __attribute__((aligned(32))) _Float16 shA[64 * 160];
    __shared__ __attribute__((aligned(32))) _Float16 shB[64 * 256];
    __shared__ int   sh_idx[NS_];
    __shared__ int   scan[256];
    __shared__ int   cntS;
    __shared__ float cc[3];

    if (tid == 0) {
        cntS = 0;
        const float* c = ctr + ((size_t)b * S + s) * 3;
        cc[0] = c[0]; cc[1] = c[1]; cc[2] = c[2];
    }
    __syncthreads();
    const float cx = cc[0], cy = cc[1], cz = cc[2];

    // ordered stream compaction of indices with d^2 <= r^2
    for (int base = 0; base < Np; base += 256) {
        __syncthreads();
        const int cbefore = cntS;
        if (cbefore >= NS_) break;
        const int i = base + tid;
        int ok = 0;
        if (i < Np) {
            const float* p = pts + ((size_t)b * Np + i) * 3;
            float dx = p[0] - cx, dy = p[1] - cy, dz = p[2] - cz;
            ok = (dx * dx + dy * dy + dz * dz) <= r2;
        }
        scan[tid] = ok;
        __syncthreads();
        for (int off = 1; off < 256; off <<= 1) {
            int v = (tid >= off) ? scan[tid - off] : 0;
            __syncthreads();
            scan[tid] += v;
            __syncthreads();
        }
        if (ok) {
            int pos = cbefore + scan[tid] - 1;
            if (pos < NS_) sh_idx[pos] = i;
        }
        __syncthreads();
        if (tid == 0) cntS = cbefore + scan[255];
    }
    __syncthreads();
    int total = cntS; if (total > NS_) total = NS_;
    if (total == 0) { if (tid == 0) sh_idx[0] = 0; total = 1; }
    __syncthreads();
    for (int k = tid; k < NS_; k += 256)
        if (k >= total) sh_idx[k] = sh_idx[0];
    __syncthreads();

    // build X0: column k holds [rel coords(3) ; feats(Cf) ; zero pad]
    for (int t = tid; t < C0pad * NS_; t += 256) {
        int k = t & 63, c = t >> 6;
        float v = 0.0f;
        if (c < 3) {
            v = pts[((size_t)b * Np + sh_idx[k]) * 3 + c] - cc[c];
        } else if (c < C0) {
            v = feats[(size_t)b * Cf * Np + (size_t)(c - 3) * Np + sh_idx[k]];
        }
        shA[k * 160 + c] = (_Float16)v;
    }
    __syncthreads();

    mlp_layer_lds(shA, 160, shB, 256, W1, b1, s1, t1, C0pad, C1, tid);
    __syncthreads();
    mlp_layer_lds(shB, 256, shA, 160, W2, b2, s2, t2, C1, C2, tid);
    __syncthreads();
    mlp_layer_lds(shA, 160, shB, 256, W3, b3, s3, t3, C2, C3, tid);
    __syncthreads();

    for (int o = tid; o < C3; o += 256) {
        float m = -3.0e38f;
        for (int k = 0; k < NS_; ++k) m = fmaxf(m, (float)shB[k * 256 + o]);
        out[(size_t)b * C3 * S + (size_t)o * S + s] = m;
    }
}

// ---------------------------------------------------------------------------
// Generic WMMA GEMM: Y(B,O,M) = epilogue(W(OxC,swizzled f16) * X(B,C,M f32)).
// Block = 8 waves, 32(O) x 64(M) tile; X K-slice staged to LDS column-major
// with f32->f16 conversion; mode 0: y+bias; mode 1: relu((y+b)*s+t).
// ---------------------------------------------------------------------------
__global__ void __launch_bounds__(256) k_gemm(
    const _Float16* __restrict__ W, const float* __restrict__ X,
    float* __restrict__ Y,
    const float* __restrict__ bias, const float* __restrict__ sc,
    const float* __restrict__ tt,
    int O, int C, int Cpad, int M, int mode)
{
    const int b  = blockIdx.z;
    const int Ob = blockIdx.y * 32;
    const int Mb = blockIdx.x * 64;
    const int tid = threadIdx.x, lane = tid & 31, wave = tid >> 5;
    const int to = wave >> 2, tm = wave & 3;
    const int nkb = Cpad >> 5;
    __shared__ __attribute__((aligned(32))) _Float16 Xs[64 * 32]; // col-major

    const float* Xb = X + (size_t)b * C * M;
    const v16h* Wv = (const v16h*)W;
    const int col = tm * 16 + (lane & 15);
    const int wm  = tid & 63;      // writer: column
    const int wk  = tid >> 6;      // writer: k-group (8 halves each)
    v8f acc = {};

    for (int kb = 0; kb < nkb; ++kb) {
        __syncthreads();
        {
            v8h pk;
            const int m = Mb + wm;
#pragma unroll
            for (int j = 0; j < 8; ++j) {
                int kk = kb * 32 + wk * 8 + j;
                pk[j] = (kk < C && m < M) ? (_Float16)Xb[(size_t)kk * M + m]
                                          : (_Float16)0.0f;
            }
            *(v8h*)(Xs + wm * 32 + wk * 8) = pk;
        }
        if ((kb + 1) * 32 < C)   // hint next K tile into caches
            __builtin_prefetch(Xb + (size_t)(kb + 1) * 32 * M + Mb, 0, 0);
        __syncthreads();
        acc = wmma_step(Wv, ((Ob >> 4) + to) * nkb + kb, Xs + col * 32, lane, acc);
    }

    const int rb = Ob + to * 16 + ((lane >= 16) ? 8 : 0);
    const int m  = Mb + col;
    if (m < M) {
#pragma unroll
        for (int r = 0; r < 8; ++r) {
            int o = rb + r;
            if (o < O) {
                float v = acc[r] + bias[o];
                if (mode == 1) v = fmaxf(v * sc[o] + tt[o], 0.0f);
                Y[(size_t)b * O * M + (size_t)o * M + m] = v;
            }
        }
    }
}

// ---------------------------------------------------------------------------
// sa3 input build: X(B,259,128) = [l2 coords(3) ; l2 feats(256)]
// ---------------------------------------------------------------------------
__global__ void k_build_sa3(const float* __restrict__ nx2,
                            const float* __restrict__ l2p,
                            float* __restrict__ X)
{
    int t = blockIdx.x * blockDim.x + threadIdx.x;
    if (t >= B_ * 259 * 128) return;
    int m = t % 128, c = (t / 128) % 259, b = t / (128 * 259);
    X[t] = (c < 3) ? nx2[((size_t)b * 128 + m) * 3 + c]
                   : l2p[((size_t)b * 256 + (c - 3)) * 128 + m];
}

// fp3 input: X(B,1280,128) = [l2 feats(256) ; broadcast l3(1024)]
__global__ void k_build_fp3(const float* __restrict__ l2p,
                            const float* __restrict__ l3p,
                            float* __restrict__ X)
{
    int t = blockIdx.x * blockDim.x + threadIdx.x;
    if (t >= B_ * 1280 * 128) return;
    int m = t % 128, c = (t / 128) % 1280, b = t / (128 * 1280);
    X[t] = (c < 256) ? l2p[((size_t)b * 256 + c) * 128 + m]
                     : l3p[(size_t)b * 1024 + (c - 256)];
}

__global__ void k_maxpool(const float* __restrict__ Y, float* __restrict__ out,
                          int M, int total)
{
    int t = blockIdx.x * blockDim.x + threadIdx.x;
    if (t >= total) return;
    const float* r = Y + (size_t)t * M;
    float m = -3.0e38f;
    for (int i = 0; i < M; ++i) m = fmaxf(m, r[i]);
    out[t] = m;
}

// 3-NN with stable (lowest-index) tie-break, matching lax.top_k
__global__ void k_knn3(const float* __restrict__ q, int Nq,
                       const float* __restrict__ src, int Ns,
                       int* __restrict__ idx3, float* __restrict__ w3)
{
    int t = blockIdx.x * blockDim.x + threadIdx.x;
    if (t >= B_ * Nq) return;
    int b = t / Nq, n = t % Nq;
    const float* Q = q + ((size_t)b * Nq + n) * 3;
    float qx = Q[0], qy = Q[1], qz = Q[2];
    float d0 = 3e38f, d1 = 3e38f, d2 = 3e38f;
    int i0 = 0, i1 = 0, i2 = 0;
    for (int j = 0; j < Ns; ++j) {
        const float* P = src + ((size_t)b * Ns + j) * 3;
        float dx = P[0] - qx, dy = P[1] - qy, dz = P[2] - qz;
        float d = dx * dx + dy * dy + dz * dz;
        if (d < d0)      { d2 = d1; i2 = i1; d1 = d0; i1 = i0; d0 = d; i0 = j; }
        else if (d < d1) { d2 = d1; i2 = i1; d1 = d; i1 = j; }
        else if (d < d2) { d2 = d; i2 = j; }
    }
    float w0 = 1.0f / (d0 + 1e-8f), w1 = 1.0f / (d1 + 1e-8f), w2 = 1.0f / (d2 + 1e-8f);
    float ws = w0 + w1 + w2;
    int base = ((size_t)b * Nq + n) * 3;
    idx3[base + 0] = i0; idx3[base + 1] = i1; idx3[base + 2] = i2;
    w3[base + 0] = w0 / ws; w3[base + 1] = w1 / ws; w3[base + 2] = w2 / ws;
}

// X(B,C1+C2,Nq) = [feats1 ; 3-NN interpolation of feats2]
__global__ void k_interp_concat(const float* __restrict__ f1, int C1v,
                                const float* __restrict__ f2, int C2v, int Ns,
                                const int* __restrict__ idx3,
                                const float* __restrict__ w3,
                                int Nq, float* __restrict__ X)
{
    const int Ctot = C1v + C2v;
    size_t t = (size_t)blockIdx.x * blockDim.x + threadIdx.x;
    size_t total = (size_t)B_ * Ctot * Nq;
    if (t >= total) return;
    int n = (int)(t % Nq);
    int c = (int)((t / Nq) % Ctot);
    int b = (int)(t / ((size_t)Nq * Ctot));
    float v;
    if (c < C1v) {
        v = f1[((size_t)b * C1v + c) * Nq + n];
    } else {
        const int*   id = idx3 + ((size_t)b * Nq + n) * 3;
        const float* w  = w3  + ((size_t)b * Nq + n) * 3;
        const float* f  = f2  + ((size_t)b * C2v + (c - C1v)) * Ns;
        v = w[0] * f[id[0]] + w[1] * f[id[1]] + w[2] * f[id[2]];
    }
    X[t] = v;
}

// log_softmax over 50 channels + transpose to (B,N,50)
__global__ void k_logsoftmax(const float* __restrict__ logits,
                             float* __restrict__ out, int total)
{
    int t = blockIdx.x * blockDim.x + threadIdx.x;
    if (t >= total) return;
    int b = t / N_, n = t % N_;
    const float* L = logits + (size_t)b * 50 * N_ + n;
    float mx = -3.0e38f;
    for (int o = 0; o < 50; ++o) mx = fmaxf(mx, L[(size_t)o * N_]);
    float se = 0.0f;
    for (int o = 0; o < 50; ++o) se += expf(L[(size_t)o * N_] - mx);
    float lse = mx + logf(se);
    float* O = out + ((size_t)b * N_ + n) * 50;
    for (int o = 0; o < 50; ++o) O[o] = L[(size_t)o * N_] - lse;
}

// ---------------------------------------------------------------------------
// Host orchestration.
// Input order (insertion order of setup_inputs): xyz, then params
// sa1(3 layers x W,b,s,t), sa2, sa3, fp3(2), fp2(2), fp1(3), head1(W,b,s,t),
// head2(W,b)  -> 71 tensors.  Workspace use: ~52 MB.
// ---------------------------------------------------------------------------
extern "C" void kernel_launch(void* const* d_in, const int* in_sizes, int n_in,
                              void* d_out, int out_size, void* d_ws, size_t ws_size,
                              hipStream_t stream)
{
    if (n_in < 71) return;
    const float* xyz = (const float*)d_in[0];
    auto P = [&](int i) { return (const float*)d_in[i]; };

    char* ws = (char*)d_ws;
    size_t off = 0;
    auto af = [&](size_t n) { float* p = (float*)(ws + off);
        off += ((n * sizeof(float) + 255) & ~(size_t)255); return p; };
    auto ah = [&](size_t n) { _Float16* p = (_Float16*)(ws + off);
        off += ((n * sizeof(_Float16) + 255) & ~(size_t)255); return p; };
    auto ai = [&](size_t n) { int* p = (int*)(ws + off);
        off += ((n * sizeof(int) + 255) & ~(size_t)255); return p; };

    float* pts0 = af((size_t)B_ * N_ * 3);
    float* nx1  = af((size_t)B_ * S1_ * 3);
    float* nx2  = af((size_t)B_ * S2_ * 3);
    float* l1p  = af((size_t)B_ * 128 * S1_);
    float* l2p  = af((size_t)B_ * 256 * S2_);
    float* l3p  = af((size_t)B_ * 1024);
    float* l2b  = af((size_t)B_ * 256 * S2_);
    float* l1b  = af((size_t)B_ * 128 * S1_);
    float* bufA = af((size_t)B_ * 128 * 2048);
    float* bufB = af((size_t)B_ * 128 * 2048);
    float* knw  = af((size_t)B_ * 2048 * 3);
    int*   kni  = ai((size_t)B_ * 2048 * 3);

    // swizzled f16 weights (zero padded to 16xO, 32xC tiles)
    const int wIn[18] = {1,5,9, 13,17,21, 25,29,33, 37,41, 45,49, 53,57,61, 65, 69};
    const int wO[18]  = {64,64,128, 128,128,256, 256,512,1024, 256,256,
                         256,128, 128,128,128, 128, 50};
    const int wC[18]  = {3,64,64, 131,128,128, 259,256,512, 1280,256,
                         384,256, 128,128,128, 128, 128};
    _Float16* Wh[18];
    for (int i = 0; i < 18; ++i) {
        int nkb = (wC[i] + 31) / 32;
        int tilesO = (wO[i] + 15) / 16;
        int total = tilesO * nkb * 512;
        Wh[i] = ah((size_t)total);
        k_repack_w<<<(total + 255) / 256, 256, 0, stream>>>(
            P(wIn[i]), Wh[i], wO[i], wC[i], nkb, total);
    }

    k_pack_xyz<<<(B_ * N_ + 255) / 256, 256, 0, stream>>>(xyz, pts0);

    // --- SA1: FPS 512, r=0.2, C 3->64->64->128
    k_fps<<<B_, 256, 0, stream>>>(pts0, nx1, N_, S1_);
    {
        dim3 g(S1_, B_);
        k_sa_fused<<<g, 256, 0, stream>>>(pts0, nullptr, nx1, N_, S1_, 0, 0.04f,
            Wh[0], P(2),  P(3),  P(4),  64,
            Wh[1], P(6),  P(7),  P(8),  64,
            Wh[2], P(10), P(11), P(12), 128,
            3, 32, l1p);
    }
    // --- SA2: FPS 128 of the 512, r=0.4, C 131->128->128->256
    k_fps<<<B_, 256, 0, stream>>>(nx1, nx2, S1_, S2_);
    {
        dim3 g(S2_, B_);
        k_sa_fused<<<g, 256, 0, stream>>>(nx1, l1p, nx2, S1_, S2_, 128, 0.16f,
            Wh[3], P(14), P(15), P(16), 128,
            Wh[4], P(18), P(19), P(20), 128,
            Wh[5], P(22), P(23), P(24), 256,
            131, 160, l2p);
    }

    auto gemm = [&](const _Float16* W, const float* X, float* Y,
                    const float* bb, const float* ss, const float* ttp,
                    int O, int C, int M, int mode) {
        int Cpad = (C + 31) & ~31;
        dim3 g((M + 63) / 64, (O + 31) / 32, B_);
        k_gemm<<<g, 256, 0, stream>>>(W, X, Y, bb, ss, ttp, O, C, Cpad, M, mode);
    };

    // --- SA3 (group_all): 259 -> 256 -> 512 -> 1024, maxpool over 128
    { int n = B_ * 259 * 128;
      k_build_sa3<<<(n + 255) / 256, 256, 0, stream>>>(nx2, l2p, bufA); }
    gemm(Wh[6], bufA, bufB, P(26), P(27), P(28), 256, 259, 128, 1);
    gemm(Wh[7], bufB, bufA, P(30), P(31), P(32), 512, 256, 128, 1);
    gemm(Wh[8], bufA, bufB, P(34), P(35), P(36), 1024, 512, 128, 1);
    { int n = B_ * 1024;
      k_maxpool<<<(n + 255) / 256, 256, 0, stream>>>(bufB, l3p, 128, n); }

    // --- FP3 (S==1 broadcast): 1280 -> 256 -> 256 on the 128 l2 points
    { int n = B_ * 1280 * 128;
      k_build_fp3<<<(n + 255) / 256, 256, 0, stream>>>(l2p, l3p, bufA); }
    gemm(Wh[9],  bufA, bufB, P(38), P(39), P(40), 256, 1280, 128, 1);
    gemm(Wh[10], bufB, l2b,  P(42), P(43), P(44), 256, 256,  128, 1);

    // --- FP2: 3-NN(512 l1 pts among 128 l2 pts), concat(128+256) -> 256 -> 128
    { int n = B_ * S1_;
      k_knn3<<<(n + 255) / 256, 256, 0, stream>>>(nx1, S1_, nx2, S2_, kni, knw); }
    { size_t n = (size_t)B_ * 384 * S1_;
      k_interp_concat<<<(unsigned)((n + 255) / 256), 256, 0, stream>>>(
          l1p, 128, l2b, 256, S2_, kni, knw, S1_, bufA); }
    gemm(Wh[11], bufA, bufB, P(46), P(47), P(48), 256, 384, S1_, 1);
    gemm(Wh[12], bufB, l1b,  P(50), P(51), P(52), 128, 256, S1_, 1);

    // --- FP1: 3-NN(2048 among 512), interp(128) -> 128 -> 128 -> 128
    { int n = B_ * N_;
      k_knn3<<<(n + 255) / 256, 256, 0, stream>>>(pts0, N_, nx1, S1_, kni, knw); }
    { size_t n = (size_t)B_ * 128 * N_;
      k_interp_concat<<<(unsigned)((n + 255) / 256), 256, 0, stream>>>(
          nullptr, 0, l1b, 128, S1_, kni, knw, N_, bufA); }
    gemm(Wh[13], bufA, bufB, P(54), P(55), P(56), 128, 128, N_, 1);
    gemm(Wh[14], bufB, bufA, P(58), P(59), P(60), 128, 128, N_, 1);
    gemm(Wh[15], bufA, bufB, P(62), P(63), P(64), 128, 128, N_, 1);

    // --- Head: feat straight into d_out region 2; logits; log_softmax
    float* feat = (float*)d_out + (size_t)B_ * N_ * 50;
    gemm(Wh[16], bufB, feat, P(66), P(67), P(68), 128, 128, N_, 1);
    gemm(Wh[17], feat, bufA, P(70), nullptr, nullptr, 50, 128, N_, 0);
    { int n = B_ * N_;
      k_logsoftmax<<<(n + 255) / 256, 256, 0, stream>>>(bufA, (float*)d_out, n); }
    (void)in_sizes; (void)out_size; (void)ws_size;
}
